// FuzzyBellLayer_43276090474768
// MI455X (gfx1250) — matrix operand order repaired and checked by reading, not compiled, not run on hardware
//
#include <hip/hip_runtime.h>
#include <hip/hip_bf16.h>

// Problem constants (match reference)
#define BDIM 4096
#define ODIM 256
#define DDIM 256
#define BROWS 64   // batch rows per workgroup

typedef __attribute__((ext_vector_type(16))) __bf16 v16bf;
typedef __attribute__((ext_vector_type(8)))  __bf16 v8bf;
typedef __attribute__((ext_vector_type(8)))  float  v8f;

union V16U { v16bf v; v8bf h[2]; };

// ---------------------------------------------------------------------------
// Kernel 1a: materialize A[o] = triu(rots,1) + triu(rots,1)^T + diag(scales)
// in bf16, layout [o][i][j] row-major over j (j is the GEMM K dim).
// ---------------------------------------------------------------------------
__global__ __launch_bounds__(256)
void build_A_bf16(const float* __restrict__ rots,
                  const float* __restrict__ scales,
                  __bf16* __restrict__ Abf)
{
    size_t idx = (size_t)blockIdx.x * 256 + threadIdx.x;   // o*D*D + i*D + j
    int j = (int)(idx & (DDIM - 1));
    int i = (int)((idx >> 8) & (DDIM - 1));
    int o = (int)(idx >> 16);
    float v;
    if (j > i)       v = rots[idx];                                      // upper
    else if (j < i)  v = rots[((size_t)o << 16) + ((size_t)j << 8) + i]; // mirror
    else             v = scales[(o << 8) + i];                           // diag
    Abf[idx] = (__bf16)v;
}

// ---------------------------------------------------------------------------
// Kernel 1b: x (f32) -> bf16
// ---------------------------------------------------------------------------
__global__ __launch_bounds__(256)
void conv_x_bf16(const float* __restrict__ x, __bf16* __restrict__ Xbf, int n)
{
    int idx = blockIdx.x * 256 + threadIdx.x;
    if (idx < n) Xbf[idx] = (__bf16)x[idx];
}

// ---------------------------------------------------------------------------
// Kernel 2: fused  Y = X @ A_o^T(+centroid) -> sum(Y^2) over D -> bell.
// Grid: (B/64, O).  Block: 256 threads = 8 waves (wave32).
// Wave w: m-pair = w>>2 (two 16-row X tiles, fragments register-resident),
//         n-tiles (w&3)*4 .. +3.
// Inner loop: double-buffered B fragment, 2 WMMAs per fragment.
// ---------------------------------------------------------------------------
__global__ __launch_bounds__(256)
void fused_gemm_bell(const __bf16* __restrict__ Abf,
                     const __bf16* __restrict__ Xbf,
                     const float*  __restrict__ centroids,   // [O*D]
                     const float*  __restrict__ bexp,        // [O]
                     float*        __restrict__ out)         // [B*O]
{
    __shared__ float rx[BROWS];          // per-row sum-of-squares accumulator

    const int o     = blockIdx.y;
    const int brow0 = blockIdx.x * BROWS;
    const int tid   = threadIdx.x;

    if (tid < BROWS) rx[tid] = 0.0f;
    __syncthreads();

    const int w     = tid >> 5;
    const int lane  = tid & 31;
    const int h     = lane >> 4;         // lane half (0/1)
    const int ln    = lane & 15;         // lane-in-half == column N
    const int mpair = w >> 2;            // 0..1 -> m-tiles {2p, 2p+1}
    const int nt0   = (w & 3) * 4;       // 4 n-tiles per wave

    // --- cache A-matrix fragments (rows of X): 2 m-tiles x 8 K-steps --------
    // ISA 16-bit A 16x32 layout: lane half 0 -> K {k0..k0+7, k0+16..k0+23}
    //                            lane half 1 -> K {k0+8..k0+15, k0+24..k0+31}
    v16bf afrag[2][8];
#pragma unroll
    for (int m = 0; m < 2; ++m) {
        const __bf16* xrow =
            Xbf + (size_t)(brow0 + (mpair * 2 + m) * 16 + ln) * DDIM;
#pragma unroll
        for (int kk = 0; kk < 8; ++kk) {
            const __bf16* p = xrow + kk * 32 + h * 8;
            V16U u;
            u.h[0] = *(const v8bf*)(p);        // K = base + h*8 .. +7
            u.h[1] = *(const v8bf*)(p + 16);   // K = base + 16 + h*8 .. +7
            afrag[m][kk] = u.v;
        }
    }

    // per-lane y^2 accumulators across this wave's n-tiles: [m][element]
    float sqacc[2][8];
#pragma unroll
    for (int m = 0; m < 2; ++m)
#pragma unroll
        for (int e = 0; e < 8; ++e) sqacc[m][e] = 0.0f;

#pragma unroll 1
    for (int t = 0; t < 4; ++t) {
        const int ntile = nt0 + t;
        const int col   = ntile * 16 + ln;     // output column i (0..255)

        // B fragment source: B[k][n] = A_o[n][k] -> contiguous row A_o[col][:]
        // ISA 16-bit B 32x16: lane half h holds K = {h*8 .. h*8+7, 16+h*8 .. 16+h*8+7}.
        const __bf16* brow = Abf + ((size_t)o * DDIM + col) * DDIM + h * 8;

        v8f acc0 = {};
        v8f acc1 = {};

        // software-pipelined (double-buffered) B fragments
        V16U bcur, bnxt;
        bcur.h[0] = *(const v8bf*)(brow);
        bcur.h[1] = *(const v8bf*)(brow + 16);
#pragma unroll
        for (int kk = 0; kk < 8; ++kk) {
            if (kk < 7) {
                const __bf16* p = brow + (kk + 1) * 32;
                bnxt.h[0] = *(const v8bf*)(p);
                bnxt.h[1] = *(const v8bf*)(p + 16);
            }
            acc0 = __builtin_amdgcn_wmma_f32_16x16x32_bf16(
                false, afrag[0][kk], false, bcur.v, (short)0, acc0, false, false);
            acc1 = __builtin_amdgcn_wmma_f32_16x16x32_bf16(
                false, afrag[1][kk], false, bcur.v, (short)0, acc1, false, false);
            bcur = bnxt;
        }

        // accumulate y^2 per lane (no cross-lane traffic in the hot loop)
        // C/D layout: element e -> row M = h*8+e, column N = ln
        const float cent = centroids[o * DDIM + col];
#pragma unroll
        for (int e = 0; e < 8; ++e) {
            float y0 = acc0[e] + cent;
            sqacc[0][e] += y0 * y0;
            float y1 = acc1[e] + cent;
            sqacc[1][e] += y1 * y1;
        }
    }

    // --- one cross-lane reduction at the end --------------------------------
#pragma unroll
    for (int m = 0; m < 2; ++m) {
#pragma unroll
        for (int e = 0; e < 8; ++e) {
            float s = sqacc[m][e];
            s += __shfl_xor(s, 1, 16);
            s += __shfl_xor(s, 2, 16);
            s += __shfl_xor(s, 4, 16);
            s += __shfl_xor(s, 8, 16);
            if (ln == 0)
                atomicAdd(&rx[(mpair * 2 + m) * 16 + h * 8 + e], s);
        }
    }
    __syncthreads();

    // rx holds ||y||^2 ; membership = 1/(1 + (rx^2)^b) since rx^(2b)=(rx^2)^b
    if (tid < BROWS) {
        float r2 = rx[tid];
        float mm = 1.0f / (1.0f + powf(r2, bexp[o]));
        out[(size_t)(brow0 + tid) * ODIM + o] = mm;
    }
}

// ---------------------------------------------------------------------------
extern "C" void kernel_launch(void* const* d_in, const int* in_sizes, int n_in,
                              void* d_out, int out_size, void* d_ws, size_t ws_size,
                              hipStream_t stream)
{
    const float* x         = (const float*)d_in[0];   // [B*D]
    const float* scales    = (const float*)d_in[1];   // [O*D]
    const float* rots      = (const float*)d_in[2];   // [O*D*D]
    const float* centroids = (const float*)d_in[3];   // [O*D]
    const float* bexp      = (const float*)d_in[4];   // [O]
    float* out             = (float*)d_out;           // [B*O]

    const int Bb = in_sizes[0] / DDIM;                // 4096

    // workspace layout: A_bf16 (O*D*D*2 = 32MB) | X_bf16 (B*D*2 = 2MB)
    const size_t abytes = (size_t)ODIM * DDIM * DDIM * sizeof(__bf16);
    const size_t xbytes = (size_t)Bb * DDIM * sizeof(__bf16);
    if (ws_size < abytes + xbytes) return;
    __bf16* Abf = (__bf16*)d_ws;
    __bf16* Xbf = (__bf16*)((char*)d_ws + abytes);

    {
        const int total = ODIM * DDIM * DDIM;
        build_A_bf16<<<total / 256, 256, 0, stream>>>(rots, scales, Abf);
    }
    {
        const int n = Bb * DDIM;
        conv_x_bf16<<<(n + 255) / 256, 256, 0, stream>>>(x, Xbf, n);
    }
    {
        dim3 grid(Bb / BROWS, ODIM);
        fused_gemm_bell<<<grid, 256, 0, stream>>>(Abf, Xbf, centroids, bexp, out);
    }
}
